// PearsonAttention_9586367005116
// MI455X (gfx1250) — matrix operand order repaired and checked by reading, not compiled, not run on hardware
//
#include <hip/hip_runtime.h>
#include <hip/hip_bf16.h>

typedef __bf16 bf16_t;
typedef __attribute__((ext_vector_type(16))) __bf16 v16bf;
typedef __attribute__((ext_vector_type(8)))  __bf16 v8bf;
typedef __attribute__((ext_vector_type(8)))  float  v8f;

#define Bb 1024
#define Nn 64
#define Ff 512
#define SA_LD  520   // bf16 stride for 64x512 tiles (+8 pad, keeps 16B align, spreads banks)
#define HID_LD 264   // bf16 stride for 64x256 hidden
#define E_LD   68    // f32 stride for 64x64 e
#define ATT_LD 72    // bf16 stride for 64x64 attention

// ---------------- WMMA fragment helpers (CDNA5 wave32 layouts) ----------------
// A (16x32 bf16): lane l holds row m=l&15; khalf=(l>>4): elem j -> K=khalf*16+j.
// One contiguous 32B chunk (two 16B loads).
__device__ __forceinline__ v16bf load_a16(const bf16_t* base, int stride, int m0, int k0, int lane) {
    const bf16_t* p = base + (m0 + (lane & 15)) * stride + k0 + ((lane >> 4) << 4);
    v8bf lo = *(const v8bf*)p;
    v8bf hi = *(const v8bf*)(p + 8);
    return __builtin_shufflevector(lo, hi, 0,1,2,3,4,5,6,7,8,9,10,11,12,13,14,15);
}
// A fragment sourced from an f32 matrix (e), converted on the fly.
__device__ __forceinline__ v16bf load_a16_f32(const float* base, int stride, int m0, int k0, int lane) {
    const float* p = base + (m0 + (lane & 15)) * stride + k0 + ((lane >> 4) << 4);
    v16bf a;
#pragma unroll
    for (int j = 0; j < 8; ++j) { a[j] = (bf16_t)p[j]; a[j + 8] = (bf16_t)p[8 + j]; }
    return a;
}
// B (32x16 bf16): lane l holds column n=l&15; K = (l>>4)*16 + j. One contiguous 32B chunk
// when the matrix is stored n-major (i.e. transposed / "weightT" layout): row n, contiguous k.
__device__ __forceinline__ v16bf load_b16(const bf16_t* base, int stride, int n0, int k0, int lane) {
    const bf16_t* p = base + (n0 + (lane & 15)) * stride + k0 + ((lane >> 4) << 4);
    v8bf lo = *(const v8bf*)p;
    v8bf hi = *(const v8bf*)(p + 8);
    return __builtin_shufflevector(lo, hi, 0,1,2,3,4,5,6,7,8,9,10,11,12,13,14,15);
}
// B fragment gathered from a K-major (row-major [K][N]) matrix: transpose on the fly.
__device__ __forceinline__ v16bf load_b16_gather(const bf16_t* base, int stride, int k0, int n0, int lane) {
    int n  = n0 + (lane & 15);
    int kh = k0 + ((lane >> 4) << 4);
    v16bf b;
#pragma unroll
    for (int j = 0; j < 16; ++j) b[j] = base[(kh + j) * stride + n];
    return b;
}
// C/D (16x16 f32): lane l holds column n=l&15, rows m = r + 8*(l>>4), r=0..7.
__device__ __forceinline__ void store_c_bf16(bf16_t* base, int stride, int m0, int n0, int lane,
                                             v8f acc, bool relu) {
    int n = n0 + (lane & 15);
    int m = m0 + ((lane >> 4) << 3);
#pragma unroll
    for (int r = 0; r < 8; ++r) {
        float v = acc[r];
        if (relu) v = fmaxf(v, 0.0f);
        base[(m + r) * stride + n] = (bf16_t)v;
    }
}

#define WMMA_BF16(a, b, c) \
    __builtin_amdgcn_wmma_f32_16x16x32_bf16(false, (a), false, (b), (short)0, (c), false, false)

// ---------------- pre-pass: convert + transpose weights to bf16 in workspace ----------------
__global__ void prep_weights(const float* __restrict__ W_h, const float* __restrict__ A1,
                             const float* __restrict__ A2, bf16_t* __restrict__ WT,
                             bf16_t* __restrict__ A1T, bf16_t* __restrict__ A2T) {
    int i = blockIdx.x * blockDim.x + threadIdx.x;
    if (i < Ff * Ff) {                       // WT[g][f] = W_h[f][g]  (512x512)
        int g = i >> 9, f = i & 511;
        WT[i] = (bf16_t)W_h[f * Ff + g];
    } else if (i < Ff * Ff + 256 * 64) {     // A1T[khid][m] = A_w1[m][khid] (256x64)
        int j = i - Ff * Ff;
        int k = j >> 6, m = j & 63;
        A1T[j] = (bf16_t)A1[m * 256 + k];
    } else if (i < Ff * Ff + 2 * 256 * 64) { // A2T[n][k] = A_w2[k][n] (64x256)
        int j = i - Ff * Ff - 256 * 64;
        int n = j >> 8, k = j & 255;
        A2T[j] = (bf16_t)A2[k * 64 + n];
    }
}

// ---------------- fused per-batch kernel ----------------
struct SMem {
    union {
        bf16_t A[Nn][SA_LD];       // h (bf16), later xm (centered Wh)
        bf16_t Hid[Nn][HID_LD];    // hidden 64x256 (reuses A after corr)
    } u;
    bf16_t Wh[Nn][SA_LD];          // Wh bf16
    float  E[Nn][E_LD];            // e, then masked logits
    bf16_t Att[Nn][ATT_LD];        // softmax(attention) bf16
    float  stat[Nn][2];            // mean, norm per row
    float  red[Nn][4];
    float  red2[Nn][4];
};

__launch_bounds__(256)
__global__ void pearson_attn_kernel(const float* __restrict__ h, const bf16_t* __restrict__ WT,
                                    const float* __restrict__ betap, const bf16_t* __restrict__ A1T,
                                    const bf16_t* __restrict__ A2T, float* __restrict__ out) {
    __shared__ SMem sm;
    const int tid  = threadIdx.x;
    const int lane = tid & 31;
    const int w    = tid >> 5;           // 8 waves
    const int b    = blockIdx.x;
    const float beta = betap[0];

    // ---- phase 0: h[b] -> bf16 LDS ----
    const float* hb = h + (size_t)b * Nn * Ff;
#pragma unroll 4
    for (int it = 0; it < 32; ++it) {
        int i4 = tid + (it << 8);        // float4 index, 8192 total
        float4 v = ((const float4*)hb)[i4];
        int e0 = i4 << 2;
        int r = e0 >> 9, k = e0 & 511;
        sm.u.A[r][k]     = (bf16_t)v.x;
        sm.u.A[r][k + 1] = (bf16_t)v.y;
        sm.u.A[r][k + 2] = (bf16_t)v.z;
        sm.u.A[r][k + 3] = (bf16_t)v.w;
    }
    __syncthreads();

    // ---- phase 1: Wh = h @ W_h   (M=64, N=512, K=512) ----
    for (int i = 0; i < 16; ++i) {
        int tt = w + (i << 3);           // 128 tiles, 16 per wave
        int mt = tt >> 5, nt = tt & 31;
        v8f acc = {};
#pragma unroll
        for (int kt = 0; kt < 16; ++kt) {
            int k0 = kt << 5;
            v16bf a  = load_a16(&sm.u.A[0][0], SA_LD, mt << 4, k0, lane);
            v16bf bb = load_b16(WT, Ff, nt << 4, k0, lane);
            acc = WMMA_BF16(a, bb, acc);
        }
        store_c_bf16(&sm.Wh[0][0], SA_LD, mt << 4, nt << 4, lane, acc, false);
    }
    __syncthreads();

    // ---- phase 2: per-row mean & centered norm ----
    {
        int r = tid >> 2, q = tid & 3;
        float s = 0.0f, s2 = 0.0f;
        const bf16_t* row = &sm.Wh[r][0];
        for (int k = q << 7; k < (q << 7) + 128; ++k) {
            float v = (float)row[k];
            s += v; s2 += v * v;
        }
        sm.red[r][q] = s; sm.red2[r][q] = s2;
    }
    __syncthreads();
    if (tid < Nn) {
        float s  = sm.red[tid][0] + sm.red[tid][1] + sm.red[tid][2] + sm.red[tid][3];
        float s2 = sm.red2[tid][0] + sm.red2[tid][1] + sm.red2[tid][2] + sm.red2[tid][3];
        float mean = s * (1.0f / 512.0f);
        float var  = s2 - 512.0f * mean * mean;
        sm.stat[tid][0] = mean;
        sm.stat[tid][1] = sqrtf(fmaxf(var, 0.0f));
    }
    __syncthreads();

    // ---- phase 3: xm = Wh - mean  (bf16, overwrites h buffer) ----
    for (int it = 0; it < 32; ++it) {
        int e0 = (tid + (it << 8)) << 2;
        int r = e0 >> 9, k = e0 & 511;
        float mu = sm.stat[r][0];
#pragma unroll
        for (int j = 0; j < 4; ++j)
            sm.u.A[r][k + j] = (bf16_t)((float)sm.Wh[r][k + j] - mu);
    }
    __syncthreads();

    // ---- phase 4: cov = xm @ xm^T, e = beta*|cov/(norm_m*norm_n+eps)| ----
    for (int i = 0; i < 2; ++i) {
        int tt = (w << 1) + i;           // 16 tiles, 2 per wave
        int mt = tt >> 2, nt = tt & 3;
        v8f acc = {};
#pragma unroll
        for (int kt = 0; kt < 16; ++kt) {
            int k0 = kt << 5;
            v16bf a  = load_a16(&sm.u.A[0][0], SA_LD, mt << 4, k0, lane);
            v16bf bb = load_b16(&sm.u.A[0][0], SA_LD, nt << 4, k0, lane); // row n, contiguous k == xm^T column
            acc = WMMA_BF16(a, bb, acc);
        }
        int n  = (nt << 4) + (lane & 15);
        int m0 = (mt << 4) + ((lane >> 4) << 3);
        float nrm_n = sm.stat[n][1];
#pragma unroll
        for (int r = 0; r < 8; ++r) {
            int m = m0 + r;
            float denom = sm.stat[m][1] * nrm_n + 1e-8f;
            sm.E[m][n] = beta * fabsf(acc[r] / denom);
        }
    }
    __syncthreads();

    // ---- phase 5: hidden = relu(e @ A_w1)  (M=64,N=256,K=64) ----
    for (int i = 0; i < 8; ++i) {
        int tt = w + (i << 3);           // 64 tiles, 8 per wave
        int mt = tt >> 4, nt = tt & 15;
        v8f acc = {};
#pragma unroll
        for (int kt = 0; kt < 2; ++kt) {
            int k0 = kt << 5;
            v16bf a  = load_a16_f32(&sm.E[0][0], E_LD, mt << 4, k0, lane);
            v16bf bb = load_b16(A1T, 64, nt << 4, k0, lane);
            acc = WMMA_BF16(a, bb, acc);
        }
        store_c_bf16(&sm.u.Hid[0][0], HID_LD, mt << 4, nt << 4, lane, acc, true);
    }
    __syncthreads();

    // ---- phase 6: adj = relu(e + hidden @ A_w2); logits = adj>0 ? e : NEG ----
    for (int i = 0; i < 2; ++i) {
        int tt = (w << 1) + i;
        int mt = tt >> 2, nt = tt & 3;
        v8f acc = {};
#pragma unroll
        for (int kt = 0; kt < 8; ++kt) {
            int k0 = kt << 5;
            v16bf a  = load_a16(&sm.u.Hid[0][0], HID_LD, mt << 4, k0, lane);
            v16bf bb = load_b16(A2T, 256, nt << 4, k0, lane);
            acc = WMMA_BF16(a, bb, acc);
        }
        int n  = (nt << 4) + (lane & 15);
        int m0 = (mt << 4) + ((lane >> 4) << 3);
#pragma unroll
        for (int r = 0; r < 8; ++r) {
            int m = m0 + r;
            float e  = sm.E[m][n];
            float adj = fmaxf(e + acc[r], 0.0f);
            sm.E[m][n] = (adj > 0.0f) ? e : -1e12f;
        }
    }
    __syncthreads();

    // ---- phase 7: row softmax -> attention bf16 ----
    if (tid < Nn) {
        float mx = -3.4e38f;
        for (int n = 0; n < Nn; ++n) mx = fmaxf(mx, sm.E[tid][n]);
        float s = 0.0f;
        for (int n = 0; n < Nn; ++n) s += __expf(sm.E[tid][n] - mx);
        float inv = 1.0f / s;
        for (int n = 0; n < Nn; ++n)
            sm.Att[tid][n] = (bf16_t)(__expf(sm.E[tid][n] - mx) * inv);
    }
    __syncthreads();

    // ---- phase 8: h_prime = attention @ Wh  (M=64,N=512,K=64) ----
    float* outb = out + (size_t)b * Nn * Ff;
    for (int i = 0; i < 16; ++i) {
        int tt = w + (i << 3);           // 128 tiles, 16 per wave
        int mt = tt >> 5, nt = tt & 31;
        v8f acc = {};
#pragma unroll
        for (int kt = 0; kt < 2; ++kt) {
            int k0 = kt << 5;
            v16bf a  = load_a16(&sm.Att[0][0], ATT_LD, mt << 4, k0, lane);
            v16bf bb = load_b16_gather(&sm.Wh[0][0], SA_LD, k0, nt << 4, lane); // transpose gather
            acc = WMMA_BF16(a, bb, acc);
        }
        int n  = (nt << 4) + (lane & 15);
        int m0 = (mt << 4) + ((lane >> 4) << 3);
#pragma unroll
        for (int r = 0; r < 8; ++r)
            outb[(size_t)(m0 + r) * Ff + n] = acc[r];
    }
}

extern "C" void kernel_launch(void* const* d_in, const int* in_sizes, int n_in,
                              void* d_out, int out_size, void* d_ws, size_t ws_size,
                              hipStream_t stream) {
    const float* h    = (const float*)d_in[0];
    const float* W_h  = (const float*)d_in[1];
    const float* beta = (const float*)d_in[2];
    const float* A_w1 = (const float*)d_in[3];
    const float* A_w2 = (const float*)d_in[4];
    float* out = (float*)d_out;

    bf16_t* WT  = (bf16_t*)d_ws;                 // 512*512 bf16
    bf16_t* A1T = WT + Ff * Ff;                  // 256*64 bf16
    bf16_t* A2T = A1T + 256 * 64;                // 64*256 bf16

    int prep_elems = Ff * Ff + 2 * 256 * 64;     // 294912
    prep_weights<<<(prep_elems + 255) / 256, 256, 0, stream>>>(W_h, A_w1, A_w2, WT, A1T, A2T);
    pearson_attn_kernel<<<Bb, 256, 0, stream>>>(h, WT, beta, A1T, A2T, out);
}
